// MultiHeadAttention_17428977287790
// MI455X (gfx1250) — compile-verified, matrix operand order checked by reading, not run on hardware
//
#include <hip/hip_runtime.h>
#include <hip/hip_bf16.h>

// ---------------------------------------------------------------------------
// MHA forward for MI455X (gfx1250, wave32, WMMA).
//   B=2, S=2048, D_MODEL=1024, H=16, Dh=64.
// Pipeline:
//   1) q = Q@Wq+bq  -> bf16 ws [B,H,S,64]
//   2) k = K@Wk+bk  -> bf16 ws [B,H,S,64]
//   3) vT= V@Wv+bv  -> bf16 ws [B,H,64,S]   (transposed for ctx B-fragments)
//   4) attention (scores computed TRANSPOSED: S^T = k @ q^T, so each lane
//      owns one q row and 8 consecutive kv cols): 2-pass exact softmax,
//      non-temporal vectorized attn stores, in-register p->A-frag assembly,
//      ctx bf16 [B,S,1024] to ws
//   5) out = ctx@Wo+bo -> fp32 d_out head (non-temporal)
// ---------------------------------------------------------------------------

typedef __bf16 bf16;
typedef __attribute__((ext_vector_type(16))) bf16  v16bf;
typedef __attribute__((ext_vector_type(8)))  bf16  v8bf;
typedef __attribute__((ext_vector_type(8)))  float v8f;
typedef __attribute__((ext_vector_type(4)))  float v4f;

constexpr int Bc = 2;
constexpr int Sc = 2048;
constexpr int Hc = 16;
constexpr int Dh = 64;
constexpr int Dm = 1024;

// A/B fragment loader for v_wmma_f32_16x16x32_bf16.
// ISA 16-bit 16x32 layout: lane = row (0..15 twice); lanes 0-15 hold
// K = 0..7 and 16..23, lanes 16-31 hold K = 8..15 and 24..31.
// p = base of this lane's row (row-major, contiguous K); kb = (lane>=16)?8:0.
__device__ inline v16bf load_frag(const bf16* p, int kb) {
  v16bf f;
#pragma unroll
  for (int i = 0; i < 8; ++i) {
    f[i]     = p[kb + i];
    f[i + 8] = p[kb + 16 + i];
  }
  return f;
}

__device__ inline v8f wmma_bf16(v16bf a, v16bf b, v8f c) {
  return __builtin_amdgcn_wmma_f32_16x16x32_bf16(false, a, false, b,
                                                 (short)0, c, false, false);
}

__device__ inline v8f v8f_zero() {
  v8f z = {0.f, 0.f, 0.f, 0.f, 0.f, 0.f, 0.f, 0.f};
  return z;
}

// ---------------------------------------------------------------------------
// GEMM: out[4096,1024] = A[4096,1024] @ W[1024,1024] + bias
// MODE 0: store bf16 at [B,H,S,64]      (q, k)
// MODE 1: store bf16 at [B,H,64,S]      (v transposed; vectorized b128 store)
// MODE 2: store fp32 at [4096,1024]     (final output, non-temporal)
// Block: 256 threads (8 waves), tile 64(M) x 128(N), K step 32.
// ---------------------------------------------------------------------------
template <int MODE>
__device__ inline void store_tile(const v8f& c, int mrow0, int ncol0,
                                  const float* __restrict__ bias,
                                  void* __restrict__ outp) {
  const int lane = threadIdx.x & 31;
  const int col = ncol0 + (lane & 15);
  const int rbase = mrow0 + ((lane >> 4) ? 8 : 0);
  const float bv = bias[col];
  if (MODE == 1) {
    // lane holds fixed col (h,d) and 8 consecutive rows (s): one b128 store
    const int b = rbase >> 11, s = rbase & (Sc - 1);
    const int h = col >> 6, d = col & (Dh - 1);
    v8bf pk;
#pragma unroll
    for (int r = 0; r < 8; ++r) pk[r] = (bf16)(c[r] + bv);
    *(v8bf*)&((bf16*)outp)[(((size_t)b * Hc + h) * Dh + d) * Sc + s] = pk;
    return;
  }
#pragma unroll
  for (int r = 0; r < 8; ++r) {
    const int m = rbase + r;
    const float v = c[r] + bv;
    if (MODE == 2) {
      __builtin_nontemporal_store(v, &((float*)outp)[(size_t)m * Dm + col]);
    } else {
      const int b = m >> 11, s = m & (Sc - 1);
      const int h = col >> 6, d = col & (Dh - 1);
      ((bf16*)outp)[(((size_t)b * Hc + h) * Sc + s) * Dh + d] = (bf16)v;
    }
  }
}

template <typename AT, int MODE>
__global__ __launch_bounds__(256) void gemm_kernel(
    const AT* __restrict__ A, const float* __restrict__ W,
    const float* __restrict__ bias, void* __restrict__ outp) {
  __shared__ bf16 sA[64 * 32];    // [m][k]
  __shared__ bf16 sB[128 * 32];   // [n][k]  (W transposed on fill)

  const int tid = threadIdx.x;
  const int lane = tid & 31, w = tid >> 5;
  const int m0 = blockIdx.y * 64, n0 = blockIdx.x * 128;
  const int wm = w >> 2, wn = w & 3;
  const int kb = (lane >> 4) * 8;
  const int row = lane & 15;

  v8f c00 = v8f_zero(), c01 = v8f_zero(), c10 = v8f_zero(), c11 = v8f_zero();

  for (int k0 = 0; k0 < Dm; k0 += 32) {
    {  // fill sA: 2048 elems, 8 contiguous per thread
      const int idx = tid * 8;
      const int r = idx >> 5, kk = idx & 31;
      const AT* src = A + (size_t)(m0 + r) * Dm + k0 + kk;
#pragma unroll
      for (int i = 0; i < 8; ++i) sA[idx + i] = (bf16)(float)src[i];
    }
    {  // fill sB transposed: thread covers one k row, 16 n values
      const int kk = tid >> 3;
      const int ns = (tid & 7) * 16;
      const float* src = W + (size_t)(k0 + kk) * Dm + n0 + ns;
#pragma unroll
      for (int i = 0; i < 16; ++i) sB[(ns + i) * 32 + kk] = (bf16)src[i];
    }
    __syncthreads();

    const bf16* pa0 = sA + (wm * 32 + row) * 32;
    const bf16* pb0 = sB + (wn * 32 + row) * 32;
    v16bf a0 = load_frag(pa0, kb);
    v16bf a1 = load_frag(pa0 + 16 * 32, kb);
    v16bf b0 = load_frag(pb0, kb);
    v16bf b1 = load_frag(pb0 + 16 * 32, kb);

    c00 = wmma_bf16(a0, b0, c00);
    c01 = wmma_bf16(a0, b1, c01);
    c10 = wmma_bf16(a1, b0, c10);
    c11 = wmma_bf16(a1, b1, c11);
    __syncthreads();
  }

  store_tile<MODE>(c00, m0 + wm * 32 + 0,  n0 + wn * 32 + 0,  bias, outp);
  store_tile<MODE>(c01, m0 + wm * 32 + 0,  n0 + wn * 32 + 16, bias, outp);
  store_tile<MODE>(c10, m0 + wm * 32 + 16, n0 + wn * 32 + 0,  bias, outp);
  store_tile<MODE>(c11, m0 + wm * 32 + 16, n0 + wn * 32 + 16, bias, outp);
}

// ---------------------------------------------------------------------------
// Attention kernel. Grid: (S/32, B*H); 256 threads (8 waves).
// Waves: wr = w>>2 picks 16-row group, wc = w&3 partitions kv columns.
// Scores computed TRANSPOSED: S^T(kv,q) = k(16x64) @ q^T(64x16).
// Transposed C layout => lane ln / 16+ln both own q row r0+ln, with the
// 8 accumulator elements being 8 CONSECUTIVE kv columns (offset 8*hi).
//   - mask: one aligned u64 load per tile
//   - attn: two b128 non-temporal stores per tile
//   - softmax stats: scalar (m,l) per lane, one shfl_xor(16) merge
//   - p A-fragment for ctx WMMA assembled entirely in registers
// ---------------------------------------------------------------------------
__global__ __launch_bounds__(256) void attn_kernel(
    const bf16* __restrict__ qws, const bf16* __restrict__ kws,
    const bf16* __restrict__ vtws, const unsigned char* __restrict__ mask,
    float* __restrict__ attn, bf16* __restrict__ ctx) {
  __shared__ float red_m[8][16];
  __shared__ float red_l[8][16];
  __shared__ float cred[8][16 * 64];    // ctx partials per wave

  const int tid = threadIdx.x, lane = tid & 31, w = tid >> 5;
  const int hi = lane >> 4, kb = hi * 8, ln = lane & 15;
  const int wr = w >> 2, wc = w & 3;
  const int bh = blockIdx.y;
  const int b = bh >> 4;
  const int q0 = blockIdx.x * 32;
  const int r0 = q0 + wr * 16;                 // wave's first q row (in S)
  const int qr = r0 + ln;                      // THIS lane's q row
  const size_t qkbase = (size_t)bh * Sc * Dh;
  const float scale = 0.125f;                  // 1/sqrt(64)

  // q^T B-fragments, loaded once (lane = q column of S^T; K = d)
  const bf16* qrow = qws + qkbase + (size_t)qr * Dh;
  const v16bf qb0 = load_frag(qrow, kb);
  const v16bf qb1 = load_frag(qrow + 32, kb);

  const unsigned char* mrow = mask + ((size_t)b * Sc + qr) * Sc;

  // ---------------- pass 1: lane-local online softmax stats ----------------
  float rm = -3.0e38f, rl = 0.f;

  for (int t = 0; t < 32; ++t) {
    const int c0 = (wc + 4 * t) * 16;
    const bf16* krow = kws + qkbase + (size_t)(c0 + ln) * Dh;
    v16bf ka0 = load_frag(krow, kb);
    v16bf ka1 = load_frag(krow + 32, kb);
    v8f acc = v8f_zero();
    acc = wmma_bf16(ka0, qb0, acc);   // S^T tile: A = k, B = q^T
    acc = wmma_bf16(ka1, qb1, acc);

    const int kvb = c0 + 8 * hi;      // first of this lane's 8 kv columns
    const unsigned long long mb = *(const unsigned long long*)(mrow + kvb);
    float sv[8], tmax = -3.0e38f;
#pragma unroll
    for (int r = 0; r < 8; ++r) {
      float s = acc[r] * scale;
      if (!((mb >> (8 * r)) & 0xffull)) s = -1e9f;
      sv[r] = s;
      tmax = fmaxf(tmax, s);
    }
    const float mn = fmaxf(rm, tmax);
    float add = 0.f;
#pragma unroll
    for (int r = 0; r < 8; ++r) add += __expf(sv[r] - mn);
    rl = rl * __expf(rm - mn) + add;
    rm = mn;
  }
  {  // merge the two 16-lane halves (they hold the same q row)
    const float mo = __shfl_xor(rm, 16, 32);
    const float lo = __shfl_xor(rl, 16, 32);
    const float mn = fmaxf(rm, mo);
    rl = rl * __expf(rm - mn) + lo * __expf(mo - mn);
    rm = mn;
  }
  if (hi == 0) {
    red_m[w][ln] = rm;
    red_l[w][ln] = rl;
  }
  __syncthreads();

  float Mf, Li;
  {
    float M = red_m[4 * wr + 0][ln];
    M = fmaxf(M, red_m[4 * wr + 1][ln]);
    M = fmaxf(M, red_m[4 * wr + 2][ln]);
    M = fmaxf(M, red_m[4 * wr + 3][ln]);
    float L = 0.f;
#pragma unroll
    for (int j = 0; j < 4; ++j)
      L += red_l[4 * wr + j][ln] * __expf(red_m[4 * wr + j][ln] - M);
    Mf = M;
    Li = (L > 0.f) ? (1.f / L) : 0.f;
  }

  // ---------------- pass 2: normalize, write attn, ctx = p @ v ----------------
  v8f cacc[4];
#pragma unroll
  for (int j = 0; j < 4; ++j) cacc[j] = v8f_zero();

  float* arow = attn + (size_t)bh * Sc * Sc + (size_t)qr * Sc;

  for (int t = 0; t < 16; ++t) {
    const int c0 = wc * 32 + t * 128;  // 32 kv columns per iteration
    v16bf pa;                          // p A-fragment, assembled in registers
#pragma unroll
    for (int sub = 0; sub < 2; ++sub) {
      const int cc = c0 + sub * 16;
      const bf16* krow = kws + qkbase + (size_t)(cc + ln) * Dh;
      v16bf ka0 = load_frag(krow, kb);
      v16bf ka1 = load_frag(krow + 32, kb);
      v8f acc = v8f_zero();
      acc = wmma_bf16(ka0, qb0, acc);
      acc = wmma_bf16(ka1, qb1, acc);

      const int kvb = cc + 8 * hi;
      const unsigned long long mb = *(const unsigned long long*)(mrow + kvb);
      v4f p0, p1;
#pragma unroll
      for (int r = 0; r < 8; ++r) {
        float s = acc[r] * scale;
        if (!((mb >> (8 * r)) & 0xffull)) s = -1e9f;
        const float p = __expf(s - Mf) * Li;
        if (r < 4) p0[r] = p; else p1[r - 4] = p;
        pa[sub * 8 + r] = (bf16)p;    // K 0..7/16..23 (lo) or 8..15/24..31 (hi)
      }
      // streaming 512MB attn output: contiguous, keep out of L2 (TH=NT)
      float* ab = arow + kvb;
      __builtin_nontemporal_store(p0, (v4f*)ab);
      __builtin_nontemporal_store(p1, (v4f*)(ab + 4));
    }
#pragma unroll
    for (int j = 0; j < 4; ++j) {
      const bf16* vrow =
          vtws + (size_t)bh * Dh * Sc + (size_t)(j * 16 + ln) * Sc + c0;
      v16bf vb = load_frag(vrow, kb);
      cacc[j] = wmma_bf16(pa, vb, cacc[j]);
    }
  }

  // cross-wave ctx reduction
#pragma unroll
  for (int j = 0; j < 4; ++j)
#pragma unroll
    for (int r = 0; r < 8; ++r)
      cred[w][(hi * 8 + r) * 64 + j * 16 + ln] = cacc[j][r];
  __syncthreads();

  {
    const int rowl = tid >> 3;          // 0..31 local q row
    const int d0 = (tid & 7) * 8;
    const int rg = rowl >> 4, rig = rowl & 15;
    const int qrow2 = q0 + rowl;
    const int h = bh & 15;
    bf16* cbase = ctx + ((size_t)b * Sc + qrow2) * Dm + h * Dh + d0;
#pragma unroll
    for (int dd = 0; dd < 8; ++dd) {
      float v = cred[4 * rg + 0][rig * 64 + d0 + dd] +
                cred[4 * rg + 1][rig * 64 + d0 + dd] +
                cred[4 * rg + 2][rig * 64 + d0 + dd] +
                cred[4 * rg + 3][rig * 64 + d0 + dd];
      cbase[dd] = (bf16)v;
    }
  }
}

// ---------------------------------------------------------------------------
extern "C" void kernel_launch(void* const* d_in, const int* in_sizes, int n_in,
                              void* d_out, int out_size, void* d_ws,
                              size_t ws_size, hipStream_t stream) {
  const float* Q = (const float*)d_in[0];
  const float* K = (const float*)d_in[1];
  const float* V = (const float*)d_in[2];
  const unsigned char* mask = (const unsigned char*)d_in[3];
  const float* Wq = (const float*)d_in[4];
  const float* bq = (const float*)d_in[5];
  const float* Wk = (const float*)d_in[6];
  const float* bk = (const float*)d_in[7];
  const float* Wv = (const float*)d_in[8];
  const float* bv = (const float*)d_in[9];
  const float* Wo = (const float*)d_in[10];
  const float* bo = (const float*)d_in[11];

  float* out = (float*)d_out;                         // [B,S,Dm]
  float* attn = out + (size_t)Bc * Sc * Dm;           // [B,H,S,S]

  const size_t qkv_elems = (size_t)Bc * Hc * Sc * Dh; // 4M bf16 each
  bf16* qws = (bf16*)d_ws;
  bf16* kws = qws + qkv_elems;
  bf16* vtws = kws + qkv_elems;
  bf16* ctxws = vtws + qkv_elems;                     // [B*S, 1024] bf16

  const dim3 gg(Dm / 128, (Bc * Sc) / 64);            // (8, 64)
  gemm_kernel<float, 0><<<gg, 256, 0, stream>>>(Q, Wq, bq, (void*)qws);
  gemm_kernel<float, 0><<<gg, 256, 0, stream>>>(K, Wk, bk, (void*)kws);
  gemm_kernel<float, 1><<<gg, 256, 0, stream>>>(V, Wv, bv, (void*)vtws);

  attn_kernel<<<dim3(Sc / 32, Bc * Hc), 256, 0, stream>>>(qws, kws, vtws, mask,
                                                          attn, ctxws);

  gemm_kernel<bf16, 2><<<gg, 256, 0, stream>>>(ctxws, Wo, bo, (void*)out);
}
